// TitansMemoryModule_20856361189879
// MI455X (gfx1250) — matrix operand order repaired and checked by reading, compiled, not run on hardware
//
#include <hip/hip_runtime.h>
#include <hip/hip_bf16.h>
#include <math.h>

// ---------------------------------------------------------------------------
// Titans memory module for MI455X (gfx1250), wave32, WMMA bf16 path.
//   retrieved = k @ mem_w^T                (16384x1024 @ 1024x1024)
//   grad      = (2/n) * (retrieved-v)^T @ k (1024x16384 @ 16384x1024)
// Both GEMMs run as v_wmma_f32_16x16x32_bf16 with fp32 accumulators.
// ---------------------------------------------------------------------------

typedef __attribute__((ext_vector_type(16))) __bf16 v16bf;
typedef __attribute__((ext_vector_type(8)))  __bf16 v8bf;
typedef __attribute__((ext_vector_type(8)))  float  v8f;

#define NB    16384      // batch rows
#define KD    1024       // key dim
#define VD    1024       // val dim
#define MAXNORM 5.0f
#define TTT_LR  0.005f

// ----------------------------- fragment loaders ----------------------------
// A-matrix 16x32 bf16 (M x K), contiguous layout (matches B):
//   lanes 0-15 : M=lane,    elems e=0..15 -> K=e
//   lanes 16-31: M=lane-16, elems e=0..15 -> K=16+e
// => lane needs one contiguous 16-bf16 chunk at K = 16*half.
__device__ __forceinline__ v16bf load_frag_a(const __bf16* __restrict__ base,
                                             int row0, int ld, int k0, int lane) {
  const int half = lane >> 4;
  const int m    = lane & 15;
  const __bf16* p = base + (size_t)(row0 + m) * (size_t)ld + (size_t)(k0 + 16 * half);
  v8bf lo = *(const v8bf*)(p);
  v8bf hi = *(const v8bf*)(p + 8);
  v16bf r;
#pragma unroll
  for (int i = 0; i < 8; ++i) { r[i] = lo[i]; r[i + 8] = hi[i]; }
  return r;
}

// B-matrix 32x16 bf16 (K x N) with B[k][n] = W[n][k], W row-major (ld = K stride).
//   lanes 0-15 : N=lane,    elems e=0..15 -> K=e
//   lanes 16-31: N=lane-16, elems e=0..15 -> K=16+e
// => lane needs one contiguous 16-bf16 chunk at K = 16*half in row n of W.
__device__ __forceinline__ v16bf load_frag_b(const __bf16* __restrict__ W,
                                             int n0, int ld, int k0, int lane) {
  const int half = lane >> 4;
  const int n    = lane & 15;
  const __bf16* p = W + (size_t)(n0 + n) * (size_t)ld + (size_t)(k0 + 16 * half);
  v8bf lo = *(const v8bf*)(p);
  v8bf hi = *(const v8bf*)(p + 8);
  v16bf r;
#pragma unroll
  for (int i = 0; i < 8; ++i) { r[i] = lo[i]; r[i + 8] = hi[i]; }
  return r;
}

__device__ __forceinline__ v8f wmma_bf16(v16bf a, v16bf b, v8f c) {
  return __builtin_amdgcn_wmma_f32_16x16x32_bf16(
      /*neg_a=*/false, a, /*neg_b=*/false, b,
      /*c_mod=*/(short)0, c, /*reuse_a=*/false, /*reuse_b=*/false);
}

// ------------------------------ small kernels ------------------------------

__global__ void init_acc_kernel(float* __restrict__ ksum, float* __restrict__ scal) {
  int t = blockIdx.x * blockDim.x + threadIdx.x;
  if (t < KD) ksum[t] = 0.0f;
  if (t < 16) scal[t] = 0.0f;
}

// k (f32) -> kb (bf16), plus per-column sums for k_agg.
__global__ void colsum_convert_kernel(const float* __restrict__ k,
                                      __bf16* __restrict__ kb,
                                      float* __restrict__ ksum) {
  const int col = blockIdx.x * 256 + threadIdx.x;
  const int r0  = blockIdx.y * 256;
  float s = 0.0f;
  for (int r = 0; r < 256; ++r) {
    float x = k[(size_t)(r0 + r) * KD + col];
    kb[(size_t)(r0 + r) * KD + col] = (__bf16)x;
    s += x;
  }
  atomicAdd(&ksum[col], s);
}

// gates = sigmoid(gate_w @ (ksum/NB) + gate_b)
__global__ void gates_kernel(const float* __restrict__ gw, const float* __restrict__ gb,
                             const float* __restrict__ ksum,
                             float* __restrict__ scal, float* __restrict__ gatesOut) {
  __shared__ float red[256];
  const int t = threadIdx.x;
  const float invB = 1.0f / (float)NB;
  for (int g = 0; g < 3; ++g) {
    float s = 0.0f;
    for (int j = t; j < KD; j += 256) s += gw[g * KD + j] * ksum[j];
    red[t] = s; __syncthreads();
    for (int st = 128; st > 0; st >>= 1) { if (t < st) red[t] += red[t + st]; __syncthreads(); }
    if (t == 0) {
      float x = red[0] * invB + gb[g];
      float gate = 1.0f / (1.0f + __expf(-x));
      scal[g] = gate;      // 0:alpha 1:eta 2:theta
      gatesOut[g] = gate;
    }
    __syncthreads();
  }
}

__global__ void convert_w_kernel(const float* __restrict__ w, __bf16* __restrict__ wb) {
  int i = blockIdx.x * 256 + threadIdx.x;
  wb[i] = (__bf16)w[i];
}

// f32 (rows x cols) -> bf16 transposed (cols x rows)
__global__ void transpose_f32_bf16_kernel(const float* __restrict__ in,
                                          __bf16* __restrict__ out,
                                          int rows, int cols) {
  __shared__ float tile[32][33];
  const int x  = blockIdx.x * 32 + threadIdx.x;   // col
  const int y0 = blockIdx.y * 32;                 // row base
  for (int j = threadIdx.y; j < 32; j += 8)
    tile[j][threadIdx.x] = in[(size_t)(y0 + j) * cols + x];
  __syncthreads();
  const int ox  = y0 + threadIdx.x;               // row index -> out col
  const int oy0 = blockIdx.x * 32;                // col base  -> out row
  for (int j = threadIdx.y; j < 32; j += 8)
    out[(size_t)(oy0 + j) * rows + ox] = (__bf16)tile[threadIdx.x][j];
}

// r = retrieved - v ; write r^T (bf16) ; accumulate sum(r^2) for loss.
__global__ void resid_transpose_kernel(const float* __restrict__ ret,
                                       const float* __restrict__ v,
                                       __bf16* __restrict__ rT,
                                       float* __restrict__ lossAcc) {
  __shared__ float tile[32][33];
  __shared__ float red[256];
  const int x  = blockIdx.x * 32 + threadIdx.x;   // col in val dim
  const int y0 = blockIdx.y * 32;                 // batch row base
  float lsum = 0.0f;
  for (int j = threadIdx.y; j < 32; j += 8) {
    size_t idx = (size_t)(y0 + j) * VD + x;
    float r = ret[idx] - v[idx];
    tile[j][threadIdx.x] = r;
    lsum += r * r;
  }
  __syncthreads();
  const int ox  = y0 + threadIdx.x;
  const int oy0 = blockIdx.x * 32;
  for (int j = threadIdx.y; j < 32; j += 8)
    rT[(size_t)(oy0 + j) * NB + ox] = (__bf16)tile[threadIdx.x][j];
  const int tid = threadIdx.y * 32 + threadIdx.x;
  red[tid] = lsum; __syncthreads();
  for (int s = 128; s > 0; s >>= 1) { if (tid < s) red[tid] += red[tid + s]; __syncthreads(); }
  if (tid == 0) atomicAdd(lossAcc, red[0]);
}

// ------------------------------- GEMM kernels ------------------------------
// 128 threads = 4 waves; block tile 64x64; each wave owns a 32x32 subtile
// (2x2 fragments of 16x16), reduction over K in steps of 32 via WMMA bf16.

__global__ void gemm1_kernel(const __bf16* __restrict__ A,   // kb  (NB x KD)
                             const __bf16* __restrict__ W,   // wb  (VD x KD)
                             float* __restrict__ outR) {     // retrieved (NB x VD)
  const int lane = threadIdx.x & 31;
  const int wv   = threadIdx.x >> 5;
  const int bm = blockIdx.y * 64 + (wv & 1) * 32;
  const int bn = blockIdx.x * 64 + (wv >> 1) * 32;
  v8f acc[2][2] = {};
  for (int k0 = 0; k0 < KD; k0 += 32) {
    if (k0 + 32 < KD)
      __builtin_prefetch(A + (size_t)(bm + (lane & 15)) * KD + k0 + 32, 0, 1);
    v16bf a0 = load_frag_a(A, bm + 0,  KD, k0, lane);
    v16bf a1 = load_frag_a(A, bm + 16, KD, k0, lane);
    v16bf b0 = load_frag_b(W, bn + 0,  KD, k0, lane);
    v16bf b1 = load_frag_b(W, bn + 16, KD, k0, lane);
    acc[0][0] = wmma_bf16(a0, b0, acc[0][0]);
    acc[0][1] = wmma_bf16(a0, b1, acc[0][1]);
    acc[1][0] = wmma_bf16(a1, b0, acc[1][0]);
    acc[1][1] = wmma_bf16(a1, b1, acc[1][1]);
  }
  const int half = lane >> 4, nn = lane & 15;
#pragma unroll
  for (int fi = 0; fi < 2; ++fi)
#pragma unroll
    for (int fj = 0; fj < 2; ++fj)
#pragma unroll
      for (int e = 0; e < 8; ++e) {
        int row = bm + 16 * fi + e + 8 * half;
        int col = bn + 16 * fj + nn;
        outR[(size_t)row * VD + col] = acc[fi][fj][e];
      }
}

// grad/momentum fused: G = rT @ kT^T over K=NB; then
//   gm = eta*momentum - TTT_LR*theta*clamp((2/n)*G, -1, 1), accumulate sum(gm^2)
__global__ void gemm2_kernel(const __bf16* __restrict__ rT,   // (VD x NB)
                             const __bf16* __restrict__ kT,   // (KD x NB)
                             const float* __restrict__ momentum,
                             const float* __restrict__ scal,
                             float* __restrict__ gm,
                             float* __restrict__ msqAcc) {
  const int lane = threadIdx.x & 31;
  const int wv   = threadIdx.x >> 5;
  const int bm = blockIdx.y * 64 + (wv & 1) * 32;   // val dim
  const int bn = blockIdx.x * 64 + (wv >> 1) * 32;  // key dim
  v8f acc[2][2] = {};
  for (int k0 = 0; k0 < NB; k0 += 32) {
    if (k0 + 32 < NB)
      __builtin_prefetch(rT + (size_t)(bm + (lane & 15)) * NB + k0 + 32, 0, 1);
    v16bf a0 = load_frag_a(rT, bm + 0,  NB, k0, lane);
    v16bf a1 = load_frag_a(rT, bm + 16, NB, k0, lane);
    v16bf b0 = load_frag_b(kT, bn + 0,  NB, k0, lane);
    v16bf b1 = load_frag_b(kT, bn + 16, NB, k0, lane);
    acc[0][0] = wmma_bf16(a0, b0, acc[0][0]);
    acc[0][1] = wmma_bf16(a0, b1, acc[0][1]);
    acc[1][0] = wmma_bf16(a1, b0, acc[1][0]);
    acc[1][1] = wmma_bf16(a1, b1, acc[1][1]);
  }
  const float eta = scal[1], theta = scal[2];
  const float gscale = 2.0f / ((float)NB * (float)VD);
  const int half = lane >> 4, nn = lane & 15;
  float local = 0.0f;
#pragma unroll
  for (int fi = 0; fi < 2; ++fi)
#pragma unroll
    for (int fj = 0; fj < 2; ++fj)
#pragma unroll
      for (int e = 0; e < 8; ++e) {
        int row = bm + 16 * fi + e + 8 * half;
        int col = bn + 16 * fj + nn;
        float g = acc[fi][fj][e] * gscale;
        g = fminf(1.0f, fmaxf(-1.0f, g));
        float m = eta * momentum[(size_t)row * KD + col] - TTT_LR * theta * g;
        gm[(size_t)row * KD + col] = m;
        local += m * m;
      }
  __shared__ float red[128];
  red[threadIdx.x] = local; __syncthreads();
  for (int s = 64; s > 0; s >>= 1) { if (threadIdx.x < s) red[threadIdx.x] += red[threadIdx.x + s]; __syncthreads(); }
  if (threadIdx.x == 0) atomicAdd(msqAcc, red[0]);
}

// ----------------------------- norm-clip chain -----------------------------

__global__ void finalize1_kernel(const float* __restrict__ scal_acc,
                                 float* __restrict__ scal,
                                 float* __restrict__ lossOut) {
  // scal_acc[8]=lossSum, scal_acc[9]=msq
  *lossOut = scal_acc[8] / ((float)NB * (float)VD);
  float n = sqrtf(scal_acc[9]);
  scal[3] = (n > MAXNORM) ? MAXNORM / (n + 1e-8f) : 1.0f;
}

__global__ void mom_weight_kernel(const float* __restrict__ gm,
                                  const float* __restrict__ mem_w,
                                  const float* __restrict__ scal,
                                  float* __restrict__ momOut,
                                  float* __restrict__ wu,
                                  float* __restrict__ wsqAcc) {
  const int idx = blockIdx.x * 256 + threadIdx.x;
  const float alpha = scal[0], s1 = scal[3];
  float nm = gm[idx] * s1;
  momOut[idx] = nm;
  float w = (1.0f - alpha) * mem_w[idx] + nm;
  wu[idx] = w;
  __shared__ float red[256];
  red[threadIdx.x] = w * w; __syncthreads();
  for (int s = 128; s > 0; s >>= 1) { if (threadIdx.x < s) red[threadIdx.x] += red[threadIdx.x + s]; __syncthreads(); }
  if (threadIdx.x == 0) atomicAdd(wsqAcc, red[0]);
}

__global__ void finalize2_kernel(const float* __restrict__ scal_acc, float* __restrict__ scal) {
  float n = sqrtf(scal_acc[10]);
  scal[4] = (n > MAXNORM) ? MAXNORM / (n + 1e-8f) : 1.0f;
}

__global__ void weight_out_kernel(const float* __restrict__ wu,
                                  const float* __restrict__ scal,
                                  float* __restrict__ wOut) {
  const int idx = blockIdx.x * 256 + threadIdx.x;
  wOut[idx] = wu[idx] * scal[4];
}

// --------------------------------- launcher --------------------------------

extern "C" void kernel_launch(void* const* d_in, const int* in_sizes, int n_in,
                              void* d_out, int out_size, void* d_ws, size_t ws_size,
                              hipStream_t stream) {
  const float* k        = (const float*)d_in[0];   // (NB, KD)
  const float* v        = (const float*)d_in[1];   // (NB, VD)
  const float* mem_w    = (const float*)d_in[2];   // (VD, KD)
  const float* gate_w   = (const float*)d_in[3];   // (3, KD)
  const float* gate_b   = (const float*)d_in[4];   // (3,)
  const float* momentum = (const float*)d_in[5];   // (VD, KD)
  float* out = (float*)d_out;

  // d_out layout: retrieved | loss | new_weight | new_momentum | gates
  const size_t RET_OFF  = 0;
  const size_t LOSS_OFF = (size_t)NB * VD;                 // 16777216
  const size_t W_OFF    = LOSS_OFF + 1;                    // 16777217
  const size_t M_OFF    = W_OFF + (size_t)VD * KD;         // 17825793
  const size_t G_OFF    = M_OFF + (size_t)VD * KD;         // 18874369

  // workspace carve-up (bytes)
  char* ws = (char*)d_ws;
  size_t o = 0;
  __bf16* kb  = (__bf16*)(ws + o); o += (size_t)NB * KD * 2;   // 32 MB
  __bf16* kT  = (__bf16*)(ws + o); o += (size_t)KD * NB * 2;   // 32 MB
  __bf16* rT  = (__bf16*)(ws + o); o += (size_t)VD * NB * 2;   // 32 MB
  __bf16* wb  = (__bf16*)(ws + o); o += (size_t)VD * KD * 2;   //  2 MB
  float*  gm  = (float*)(ws + o);  o += (size_t)VD * KD * 4;   //  4 MB
  float*  wu  = (float*)(ws + o);  o += (size_t)VD * KD * 4;   //  4 MB
  float*  ksum = (float*)(ws + o); o += KD * 4;
  float*  scal = (float*)(ws + o); o += 64;   // [0..2]=gates [3]=scale1 [4]=scale2
                                              // [8]=lossSum [9]=msq [10]=wsq

  // 0) zero accumulators
  hipLaunchKernelGGL(init_acc_kernel, dim3(1), dim3(1024), 0, stream, ksum, scal);
  // 1) k -> bf16 + column sums
  hipLaunchKernelGGL(colsum_convert_kernel, dim3(KD / 256, NB / 256), dim3(256), 0, stream,
                     k, kb, ksum);
  // 2) gates
  hipLaunchKernelGGL(gates_kernel, dim3(1), dim3(256), 0, stream,
                     gate_w, gate_b, ksum, scal, out + G_OFF);
  // 3) mem_w -> bf16
  hipLaunchKernelGGL(convert_w_kernel, dim3((VD * KD) / 256), dim3(256), 0, stream, mem_w, wb);
  // 4) k -> kT (bf16)
  hipLaunchKernelGGL(transpose_f32_bf16_kernel, dim3(KD / 32, NB / 32), dim3(32, 8), 0, stream,
                     k, kT, NB, KD);
  // 5) GEMM1: retrieved = kb @ wb^T  (WMMA bf16)
  hipLaunchKernelGGL(gemm1_kernel, dim3(VD / 64, NB / 64), dim3(128), 0, stream,
                     kb, wb, out + RET_OFF);
  // 6) residual transpose + loss partials
  hipLaunchKernelGGL(resid_transpose_kernel, dim3(VD / 32, NB / 32), dim3(32, 8), 0, stream,
                     out + RET_OFF, v, rT, scal + 8);
  // 7) GEMM2 (WMMA bf16, K=16384) fused with clamp + momentum pre-update + sum(m^2)
  hipLaunchKernelGGL(gemm2_kernel, dim3(KD / 64, VD / 64), dim3(128), 0, stream,
                     rT, kT, momentum, scal, gm, scal + 9);
  // 8) loss out + momentum norm-clip scale
  hipLaunchKernelGGL(finalize1_kernel, dim3(1), dim3(1), 0, stream, scal, scal, out + LOSS_OFF);
  // 9) new_momentum out + weight pre-update + sum(w^2)
  hipLaunchKernelGGL(mom_weight_kernel, dim3((VD * KD) / 256), dim3(256), 0, stream,
                     gm, mem_w, scal, out + M_OFF, wu, scal + 10);
  // 10) weight norm-clip scale
  hipLaunchKernelGGL(finalize2_kernel, dim3(1), dim3(1), 0, stream, scal, scal);
  // 11) new_weight out
  hipLaunchKernelGGL(weight_out_kernel, dim3((VD * KD) / 256), dim3(256), 0, stream,
                     wu, scal, out + W_OFF);
}
